// CustomMultiHeadAttention_43499428774240
// MI455X (gfx1250) — matrix operand order repaired and checked by reading, not compiled, not run on hardware
//
#include <hip/hip_runtime.h>

#define B_  2
#define N_  2048
#define E_  768
#define H_  12
#define HD_ 64
#define E3_ (3 * E_)    // 2304
#define BN_ (B_ * N_)   // 4096
#define BH_ (B_ * H_)   // 24

#define SPAD_ 2052          // LDS score row stride (floats); 8*SPAD_ % 64 != 0 -> conflict-free
#define SMEM_FLOATS_ (16 * SPAD_ + 8 * 1024)
#define SMEM_BYTES_  (SMEM_FLOATS_ * 4)   // 164,096 B dynamic LDS

typedef __attribute__((ext_vector_type(16))) __bf16 v16bf;
typedef __attribute__((ext_vector_type(2)))  __bf16 bf16x2;
typedef __attribute__((ext_vector_type(8)))  float  v8f;

// ---------------------------------------------------------------------------
// WMMA fragment loaders (contiguous k order shared by A and B, wave32)
// A (16x32 bf16): lane L -> row M = L&15;
//   regs 0..7 hold K = kb+0..15 contiguous, kb = (L>>4)*16
// ---------------------------------------------------------------------------
__device__ __forceinline__ v16bf load_frag_a_bf16(const __bf16* __restrict__ base, int stride,
                                                  int row, int k0, int lane) {
  const __bf16* p = base + (size_t)(row + (lane & 15)) * stride + k0 + ((lane >> 4) << 4);
  v16bf f;
#pragma unroll
  for (int r = 0; r < 8; ++r) {
    bf16x2 q = *(const bf16x2*)(p + (r << 1));
    f[2 * r]     = q.x;
    f[2 * r + 1] = q.y;
  }
  return f;
}

// Same A layout, fp32 source (LDS or global), convert to bf16 on the fly
__device__ __forceinline__ v16bf load_frag_a_f32(const float* base, int stride,
                                                 int row, int k0, int lane) {
  const float* p = base + (size_t)(row + (lane & 15)) * stride + k0 + ((lane >> 4) << 4);
  v16bf f;
#pragma unroll
  for (int r = 0; r < 8; ++r) {
    float2 q = *(const float2*)(p + (r << 1));
    f[2 * r]     = (__bf16)q.x;
    f[2 * r + 1] = (__bf16)q.y;
  }
  return f;
}

// B (32x16 bf16) loaded from B^T stored row-major [n][k]:
//   lane L -> col N = L&15; K = (L>>4)*16 + e, e contiguous -> one 32B load
__device__ __forceinline__ v16bf load_frag_bt_bf16(const __bf16* __restrict__ base, int stride,
                                                   int ncol, int k0, int lane) {
  const __bf16* p = base + (size_t)(ncol + (lane & 15)) * stride + k0 + ((lane >> 4) << 4);
  return *(const v16bf*)p;
}

// ---------------------------------------------------------------------------
// Generic GEMM: C[M,Ncols] = A[M,K](bf16) @ Bt[Ncols,K](bf16, = B^T) + bias
// One wave -> 16x64 tile (4 WMMAs / K-step, A fragment reused)
// ---------------------------------------------------------------------------
__global__ __launch_bounds__(256) void gemm_bf16_kernel(
    const __bf16* __restrict__ A, const __bf16* __restrict__ Bt,
    const float* __restrict__ bias, float* __restrict__ C,
    int M, int Ncols, int K) {
  const int lane = threadIdx.x & 31;
  const int wid  = (int)((blockIdx.x * blockDim.x + threadIdx.x) >> 5);
  const int tilesN = Ncols >> 6;
  const int tilesM = M >> 4;
  if (wid >= tilesM * tilesN) return;
  const int tileM = (wid / tilesN) << 4;
  const int tileN = (wid % tilesN) << 6;

  v8f acc[4] = {};
  for (int k0 = 0; k0 < K; k0 += 32) {
    if (k0 + 32 < K)
      __builtin_prefetch(A + (size_t)(tileM + (lane & 15)) * K + k0 + 32, 0, 0);
    v16bf a = load_frag_a_bf16(A, K, tileM, k0, lane);
#pragma unroll
    for (int j = 0; j < 4; ++j) {
      v16bf b = load_frag_bt_bf16(Bt, K, tileN + (j << 4), k0, lane);
      acc[j] = __builtin_amdgcn_wmma_f32_16x16x32_bf16(false, a, false, b,
                                                       (short)0, acc[j], false, false);
    }
  }
  const int rowBase = tileM + ((lane >> 4) << 3);
  const int col     = tileN + (lane & 15);
#pragma unroll
  for (int j = 0; j < 4; ++j) {
    const float bv = bias[col + (j << 4)];
#pragma unroll
    for (int r = 0; r < 8; ++r)
      C[(size_t)(rowBase + r) * Ncols + col + (j << 4)] = acc[j][r] + bv;
  }
}

// ---------------------------------------------------------------------------
// Fused attention: one block = one (b,h) + 16 query rows.
//  stage 1: S = scale * Q K^T  (8 waves x 256 key-cols each) -> LDS fp32
//  stage 2: row softmax in LDS; write normalized probs to global ONCE
//  stage 3: P @ V from LDS (waves split key range; LDS partial reduction)
// ---------------------------------------------------------------------------
__global__ __launch_bounds__(256) void fused_attn_kernel(
    const __bf16* __restrict__ qkv_bf, const __bf16* __restrict__ vt,
    float* __restrict__ probs, float* __restrict__ attn_out) {
  extern __shared__ float smem[];
  float* sc   = smem;                 // [16][SPAD_] score strip
  float* part = smem + 16 * SPAD_;    // [8][1024] PV partials
  __shared__ float red[256];

  const int tid  = threadIdx.x;
  const int lane = tid & 31;
  const int w    = tid >> 5;          // wave 0..7
  const int bh    = blockIdx.x >> 7;          // N_/16 = 128 strips per head
  const int tileM = (blockIdx.x & 127) << 4;
  const int b = bh / H_;
  const int h = bh % H_;

  const __bf16* qbase = qkv_bf + (size_t)b * N_ * E3_ + h * HD_;
  const __bf16* kbase = qkv_bf + (size_t)b * N_ * E3_ + E_ + h * HD_;
  const __bf16* vbase = vt + (size_t)bh * HD_ * N_;

  // ---- stage 1: scores for cols [w*256, w*256+256) -------------------------
  {
    v8f acc[16];
#pragma unroll
    for (int jt = 0; jt < 16; ++jt) acc[jt] = (v8f){};
#pragma unroll
    for (int k0 = 0; k0 < HD_; k0 += 32) {
      v16bf a = load_frag_a_bf16(qbase, E3_, tileM, k0, lane);
#pragma unroll
      for (int jt = 0; jt < 16; ++jt) {
        v16bf bf = load_frag_bt_bf16(kbase, E3_, (w << 8) + (jt << 4), k0, lane);
        acc[jt] = __builtin_amdgcn_wmma_f32_16x16x32_bf16(false, a, false, bf,
                                                          (short)0, acc[jt], false, false);
      }
    }
    const float scale = 0.125f;  // 1/sqrt(64)
    const int rbase = (lane >> 4) << 3;
#pragma unroll
    for (int jt = 0; jt < 16; ++jt) {
      const int col = (w << 8) + (jt << 4) + (lane & 15);
#pragma unroll
      for (int r = 0; r < 8; ++r)
        sc[(rbase + r) * SPAD_ + col] = acc[jt][r] * scale;
    }
  }
  __syncthreads();

  // ---- stage 2: softmax per row (16 threads per row, 128 cols/thread) ------
  {
    const int row = tid >> 4;
    const int c0  = tid & 15;
    float* srow = sc + row * SPAD_;
    float m = -__builtin_inff();
#pragma unroll 8
    for (int i = 0; i < 128; ++i) m = fmaxf(m, srow[c0 + (i << 4)]);
    red[tid] = m;
    __syncthreads();
#pragma unroll
    for (int s = 8; s > 0; s >>= 1) {
      if ((tid & 15) < s) red[tid] = fmaxf(red[tid], red[tid + s]);
      __syncthreads();
    }
    m = red[tid & ~15];
    __syncthreads();
    float sum = 0.f;
#pragma unroll 8
    for (int i = 0; i < 128; ++i) {
      const int c = c0 + (i << 4);
      float e = __expf(srow[c] - m);
      srow[c] = e;
      sum += e;
    }
    red[tid] = sum;
    __syncthreads();
#pragma unroll
    for (int s = 8; s > 0; s >>= 1) {
      if ((tid & 15) < s) red[tid] += red[tid + s];
      __syncthreads();
    }
    const float inv = 1.f / red[tid & ~15];
    float* prow = probs + ((size_t)bh * N_ + tileM + row) * N_;
#pragma unroll 8
    for (int i = 0; i < 128; ++i) {
      const int c = c0 + (i << 4);
      float pn = srow[c] * inv;
      srow[c] = pn;        // normalized probs stay in LDS for PV
      prow[c] = pn;        // single global write of attn_probs
    }
  }
  __syncthreads();

  // ---- stage 3: P @ V, wave w covers keys [w*256, w*256+256) ---------------
  {
    v8f acc[4] = {};
    const int kBeg = w << 8;
#pragma unroll 2
    for (int k0 = kBeg; k0 < kBeg + 256; k0 += 32) {
      v16bf a = load_frag_a_f32(sc, SPAD_, 0, k0, lane);
#pragma unroll
      for (int j = 0; j < 4; ++j) {
        v16bf bf = load_frag_bt_bf16(vbase, N_, (j << 4), k0, lane);
        acc[j] = __builtin_amdgcn_wmma_f32_16x16x32_bf16(false, a, false, bf,
                                                         (short)0, acc[j], false, false);
      }
    }
    const int rbase = (lane >> 4) << 3;
#pragma unroll
    for (int j = 0; j < 4; ++j) {
      const int col = (j << 4) + (lane & 15);
#pragma unroll
      for (int r = 0; r < 8; ++r)
        part[(w << 10) + ((rbase + r) << 6) + col] = acc[j][r];
    }
  }
  __syncthreads();

  // reduce 8 wave-partials -> attn_out
  for (int e = tid; e < 1024; e += 256) {
    float s = 0.f;
#pragma unroll
    for (int ww = 0; ww < 8; ++ww) s += part[(ww << 10) + e];
    const int row = e >> 6;
    const int d   = e & 63;
    attn_out[(size_t)(b * N_ + tileM + row) * E_ + h * HD_ + d] = s;
  }
}

// ---------------------------------------------------------------------------
// Helpers: fp32 -> bf16 convert / transpose-convert / build transposed V
// ---------------------------------------------------------------------------
__global__ __launch_bounds__(256) void cvt_f32_to_bf16(const float* __restrict__ src,
                                                       __bf16* __restrict__ dst, long n) {
  long i = (long)blockIdx.x * blockDim.x + threadIdx.x;
  if (i < n) dst[i] = (__bf16)src[i];
}

__global__ __launch_bounds__(256) void transpose_cvt_kernel(const float* __restrict__ src,
                                                            __bf16* __restrict__ dst,
                                                            int R, int Ccols) {
  long i = (long)blockIdx.x * blockDim.x + threadIdx.x;
  if (i >= (long)R * Ccols) return;
  int r = (int)(i % R);
  int c = (int)(i / R);
  dst[(size_t)c * R + r] = (__bf16)src[(size_t)r * Ccols + c];
}

__global__ __launch_bounds__(256) void vt_build_kernel(const float* __restrict__ qkv,
                                                       __bf16* __restrict__ vt) {
  long i = (long)blockIdx.x * blockDim.x + threadIdx.x;
  if (i >= (long)BH_ * HD_ * N_) return;
  int  n  = (int)(i % N_);
  long r  = i / N_;
  int  d  = (int)(r % HD_);
  int  bh = (int)(r / HD_);
  int  b  = bh / H_;
  int  h  = bh % H_;
  vt[i] = (__bf16)qkv[(size_t)(b * N_ + n) * E3_ + 2 * E_ + h * HD_ + d];
}

// ---------------------------------------------------------------------------
extern "C" void kernel_launch(void* const* d_in, const int* in_sizes, int n_in,
                              void* d_out, int out_size, void* d_ws, size_t ws_size,
                              hipStream_t stream) {
  (void)in_sizes; (void)n_in; (void)out_size; (void)ws_size;
  const float* x     = (const float*)d_in[0];
  const float* w_qkv = (const float*)d_in[1];
  const float* b_qkv = (const float*)d_in[2];
  const float* w_out = (const float*)d_in[3];
  const float* b_out = (const float*)d_in[4];

  float* out   = (float*)d_out;              // [B,N,E]
  float* probs = out + (size_t)BN_ * E_;     // [B,H,N,N]

  char* ws = (char*)d_ws;
  size_t off = 0;
  auto walloc = [&](size_t bytes) -> void* {
    void* p = ws + off;
    off += (bytes + 255) & ~(size_t)255;
    return p;
  };
  float*  qkv_f32 = (float*) walloc((size_t)BN_ * E3_ * 4);        // 37.7 MB
  __bf16* x_bf    = (__bf16*)walloc((size_t)BN_ * E_ * 2);         //  6.3 MB
  __bf16* wqkv_t  = (__bf16*)walloc((size_t)E3_ * E_ * 2);         //  3.5 MB
  __bf16* wout_t  = (__bf16*)walloc((size_t)E_ * E_ * 2);          //  1.2 MB
  __bf16* qkv_bf  = (__bf16*)walloc((size_t)BN_ * E3_ * 2);        // 18.9 MB
  __bf16* vtb     = (__bf16*)walloc((size_t)BH_ * HD_ * N_ * 2);   //  6.3 MB
  float*  attn_o  = (float*) walloc((size_t)BN_ * E_ * 4);         // 12.6 MB
  __bf16* attn_bf = (__bf16*)walloc((size_t)BN_ * E_ * 2);         //  6.3 MB

  auto nb = [](long t) { return (unsigned)((t + 255) / 256); };

  // 1) operand conversions
  cvt_f32_to_bf16<<<nb((long)BN_ * E_), 256, 0, stream>>>(x, x_bf, (long)BN_ * E_);
  transpose_cvt_kernel<<<nb((long)E_ * E3_), 256, 0, stream>>>(w_qkv, wqkv_t, E_, E3_);
  transpose_cvt_kernel<<<nb((long)E_ * E_), 256, 0, stream>>>(w_out, wout_t, E_, E_);

  // 2) qkv = x @ w_qkv + b_qkv
  long w1 = (long)(BN_ / 16) * (E3_ / 64) * 32;
  gemm_bf16_kernel<<<nb(w1), 256, 0, stream>>>(x_bf, wqkv_t, b_qkv, qkv_f32, BN_, E3_, E_);

  // 3) bf16 copies for attention operands
  cvt_f32_to_bf16<<<nb((long)BN_ * E3_), 256, 0, stream>>>(qkv_f32, qkv_bf, (long)BN_ * E3_);
  vt_build_kernel<<<nb((long)BH_ * HD_ * N_), 256, 0, stream>>>(qkv_f32, vtb);

  // 4) fused scores + softmax + PV (probs written to global exactly once)
  (void)hipFuncSetAttribute((const void*)fused_attn_kernel,
                            hipFuncAttributeMaxDynamicSharedMemorySize, SMEM_BYTES_);
  fused_attn_kernel<<<BH_ * (N_ / 16), 256, SMEM_BYTES_, stream>>>(qkv_bf, vtb, probs, attn_o);

  // 5) out = attn_out @ w_out + b_out
  cvt_f32_to_bf16<<<nb((long)BN_ * E_), 256, 0, stream>>>(attn_o, attn_bf, (long)BN_ * E_);
  long w4 = (long)(BN_ / 16) * (E_ / 64) * 32;
  gemm_bf16_kernel<<<nb(w4), 256, 0, stream>>>(attn_bf, wout_t, b_out, out, BN_, E_, E_);
}
